// GCN_20547123544817
// MI455X (gfx1250) — matrix operand order repaired and checked by reading, not compiled, not run on hardware
//
#include <hip/hip_runtime.h>

// ---------------------------------------------------------------------------
// GCN forward for MI455X (gfx1250, wave32, WMMA).
//   agg   = mean-normalized neighbor aggregation (atomic segment sums, L2-resident)
//   h1    = relu(agg @ Wc^T + bc)   [100000,128]x[128,256]  (f16 WMMA, f32 acc)
//   h2    = relu(h1  @ W1^T + b1)   [100000,256]x[256,128]
//   h3    = relu(h2  @ W2^T + b2)   [100000,128]x[128,64]
//   out   = h3 @ W3^T + b3          [100000,64]x[64,1]      (dot kernel)
//
// GEMM: each wave32 owns a 16x64 output strip (4 x 16x16 WMMA accumulators),
// so the A fragment is loaded once per K-step and fed to 4 v_wmma ops.
// ---------------------------------------------------------------------------

#define N_NODES 100000
#define N_EDGES 800000
#define N_IN    128
#define N_H1    256
#define N_H2    128
#define N_H3    64

typedef _Float16 h4   __attribute__((ext_vector_type(4)));
typedef _Float16 h8   __attribute__((ext_vector_type(8)));
typedef _Float16 h16  __attribute__((ext_vector_type(16)));
typedef float    f8   __attribute__((ext_vector_type(8)));

// ------------------------------- helpers -----------------------------------

__global__ void k_zero_f32(float* __restrict__ p, int n) {
    int i = blockIdx.x * blockDim.x + threadIdx.x;
    if (i < n) p[i] = 0.0f;
}

// deg/cnt start at 1.0 to account for self loops
__global__ void k_fill1(float* __restrict__ a, float* __restrict__ b, int n) {
    int i = blockIdx.x * blockDim.x + threadIdx.x;
    if (i < n) { a[i] = 1.0f; b[i] = 1.0f; }
}

__global__ void k_cvt_f32_f16(const float* __restrict__ src,
                              _Float16* __restrict__ dst, int n) {
    int i = blockIdx.x * blockDim.x + threadIdx.x;
    if (i < n) dst[i] = (_Float16)src[i];
}

// ------------------------- aggregation kernels -----------------------------

// deg[row[e]] += 1 ; cnt[col[e]] += 1   (edge-parallel, f32 atomics in L2)
__global__ void k_degcnt(const long long* __restrict__ row,
                         const long long* __restrict__ col,
                         float* __restrict__ deg, float* __restrict__ cnt) {
    int e = blockIdx.x * blockDim.x + threadIdx.x;
    if (e < N_EDGES) {
        atomicAdd(&deg[(int)row[e]], 1.0f);
        atomicAdd(&cnt[(int)col[e]], 1.0f);
    }
}

// s[col[e]] += (1/deg[row[e]]) * x[row[e]]   one wave32 per edge, 4 floats/lane
__global__ void k_msg(const float* __restrict__ x,
                      const long long* __restrict__ row,
                      const long long* __restrict__ col,
                      const float* __restrict__ deg,
                      float* __restrict__ s) {
    int gid  = blockIdx.x * blockDim.x + threadIdx.x;
    int e    = gid >> 5;
    int lane = gid & 31;
    if (e >= N_EDGES) return;
    int rs = (int)row[e];
    int cd = (int)col[e];
    float norm = 1.0f / deg[rs];               // deg >= 1 after self loops
    const float4 xv = *(const float4*)(x + (size_t)rs * N_IN + lane * 4);
    float* sp = s + (size_t)cd * N_IN + lane * 4;
    atomicAdd(sp + 0, norm * xv.x);
    atomicAdd(sp + 1, norm * xv.y);
    atomicAdd(sp + 2, norm * xv.z);
    atomicAdd(sp + 3, norm * xv.w);
}

// agg = (s + x/deg) / cnt, stored as f16 for WMMA consumption (4 feats/thread)
__global__ void k_finalize(const float* __restrict__ s,
                           const float* __restrict__ x,
                           const float* __restrict__ deg,
                           const float* __restrict__ cnt,
                           _Float16* __restrict__ aggh) {
    int idx = blockIdx.x * blockDim.x + threadIdx.x;        // N_NODES*32 threads
    if (idx >= N_NODES * (N_IN / 4)) return;
    int i  = idx >> 5;                                      // node
    int f  = (idx & 31) * 4;                                // feature base
    const float4 sv = *(const float4*)(s + (size_t)i * N_IN + f);
    const float4 xv = *(const float4*)(x + (size_t)i * N_IN + f);
    float di = 1.0f / deg[i];
    float ci = 1.0f / cnt[i];
    h4 o;
    o[0] = (_Float16)((sv.x + xv.x * di) * ci);
    o[1] = (_Float16)((sv.y + xv.y * di) * ci);
    o[2] = (_Float16)((sv.z + xv.z * di) * ci);
    o[3] = (_Float16)((sv.w + xv.w * di) * ci);
    *(h4*)(aggh + (size_t)i * N_IN + f) = o;
}

// ----------------------------- WMMA GEMM -----------------------------------
// Y[N,O] = act(A[N,K] @ W[O,K]^T + bias), A/W f16 row-major, f32 accumulate.
// Each wave computes a 16x64 strip: 4 accumulators, A loaded once per K-step.
// Waves per block = O/64 (layer1: 4, layer2: 2, layer3: 1); grid.x = M/16.
// EXEC all-ones (no divergence), as WMMA requires.
//
// A fragment (16-bit A 16x32): lane<16 holds row M=lane,
// halves 0..15 = K 0..15; lane>=16 holds K 16..31 (matching B)
// -> two aligned 16B loads + shuffle.
// B fragment (32x16): lane<16 holds col N=lane, K 0..15; lane>=16 holds
// col N=lane-16, K 16..31 -> one contiguous 32B load per W row.
// C/D: lane<16 -> (M = j, N = lane); lane>=16 -> (M = 8+j, N = lane-16).
template <int K, int O, bool RELU>
__global__ __launch_bounds__(128)
void k_wmma_gemm(const _Float16* __restrict__ A,
                 const _Float16* __restrict__ W,
                 const float* __restrict__ bias,
                 _Float16* __restrict__ Y) {
    const int lane   = threadIdx.x & 31;
    const int wave   = threadIdx.x >> 5;       // 0 .. O/64-1
    const int hid    = lane >> 4;              // half-wave id
    const int r      = lane & 15;
    const int mBase  = blockIdx.x * 16;
    const int nGroup = wave * 64;              // 64-column strip for this wave
    const int arow   = mBase + r;              // A row this lane loads

    f8 c0 = {}, c1 = {}, c2 = {}, c3 = {};
#pragma unroll
    for (int kb = 0; kb < K; kb += 32) {
        const h8 alo = *(const h8*)(A + (size_t)arow * K + kb + hid * 16);
        const h8 ahi = *(const h8*)(A + (size_t)arow * K + kb + hid * 16 + 8);
        h16 a = __builtin_shufflevector(alo, ahi,
                                        0, 1, 2, 3, 4, 5, 6, 7,
                                        8, 9, 10, 11, 12, 13, 14, 15);
        const _Float16* wp = W + (size_t)(nGroup + r) * K + kb + hid * 16;
        h16 b0 = *(const h16*)(wp);
        h16 b1 = *(const h16*)(wp + (size_t)16 * K);
        h16 b2 = *(const h16*)(wp + (size_t)32 * K);
        h16 b3 = *(const h16*)(wp + (size_t)48 * K);
        c0 = __builtin_amdgcn_wmma_f32_16x16x32_f16(false, a, false, b0, (short)0, c0, false, false);
        c1 = __builtin_amdgcn_wmma_f32_16x16x32_f16(false, a, false, b1, (short)0, c1, false, false);
        c2 = __builtin_amdgcn_wmma_f32_16x16x32_f16(false, a, false, b2, (short)0, c2, false, false);
        c3 = __builtin_amdgcn_wmma_f32_16x16x32_f16(false, a, false, b3, (short)0, c3, false, false);
    }

    f8 acc[4] = {c0, c1, c2, c3};
#pragma unroll
    for (int t = 0; t < 4; ++t) {
        const int nCol = nGroup + t * 16 + r;  // this lane's output column
        const float bn = bias[nCol];
#pragma unroll
        for (int j = 0; j < 8; ++j) {
            int m = mBase + hid * 8 + j;
            float v = acc[t][j] + bn;
            if (RELU) v = v > 0.0f ? v : 0.0f;
            Y[(size_t)m * O + nCol] = (_Float16)v;
        }
    }
}

// --------------------------- final [64] -> 1 -------------------------------

__global__ void k_final(const _Float16* __restrict__ h3,
                        const float* __restrict__ W3,
                        const float* __restrict__ b3,
                        float* __restrict__ out) {
    int m = blockIdx.x * blockDim.x + threadIdx.x;
    if (m >= N_NODES) return;
    const _Float16* rp = h3 + (size_t)m * N_H3;
    float acc = b3[0];
#pragma unroll
    for (int j = 0; j < N_H3; ++j) acc += (float)rp[j] * W3[j];
    out[m] = acc;
}

// ------------------------------- launch ------------------------------------

extern "C" void kernel_launch(void* const* d_in, const int* in_sizes, int n_in,
                              void* d_out, int out_size, void* d_ws, size_t ws_size,
                              hipStream_t stream) {
    (void)in_sizes; (void)n_in; (void)out_size; (void)ws_size;

    const float*     x   = (const float*)d_in[0];
    const long long* ei  = (const long long*)d_in[1];   // int64 [2, E]
    const float*     Wc  = (const float*)d_in[2];
    const float*     bc  = (const float*)d_in[3];
    const float*     W1  = (const float*)d_in[4];
    const float*     b1  = (const float*)d_in[5];
    const float*     W2  = (const float*)d_in[6];
    const float*     b2  = (const float*)d_in[7];
    const float*     W3  = (const float*)d_in[8];
    const float*     b3  = (const float*)d_in[9];
    float*           out = (float*)d_out;

    const long long* row = ei;             // edge_index[0]
    const long long* col = ei + N_EDGES;   // edge_index[1]

    // workspace carve-out (256B aligned slices)
    char* p = (char*)d_ws;
    auto carve = [&](size_t bytes) -> char* {
        char* r = p;
        p += (bytes + 255) & ~(size_t)255;
        return r;
    };
    float*    deg  = (float*)   carve((size_t)N_NODES * 4);
    float*    cnt  = (float*)   carve((size_t)N_NODES * 4);
    float*    s    = (float*)   carve((size_t)N_NODES * N_IN * 4);
    _Float16* aggh = (_Float16*)carve((size_t)N_NODES * N_IN * 2);
    _Float16* h1   = (_Float16*)carve((size_t)N_NODES * N_H1 * 2);
    _Float16* h2   = (_Float16*)carve((size_t)N_NODES * N_H2 * 2);
    _Float16* h3   = (_Float16*)carve((size_t)N_NODES * N_H3 * 2);
    _Float16* Wch  = (_Float16*)carve((size_t)N_H1 * N_IN * 2);
    _Float16* W1h  = (_Float16*)carve((size_t)N_H2 * N_H1 * 2);
    _Float16* W2h  = (_Float16*)carve((size_t)N_H3 * N_H2 * 2);

    const int B = 256;
    auto blocks = [](long long n, int b) { return (int)((n + b - 1) / b); };

    // init
    k_zero_f32<<<blocks((long long)N_NODES * N_IN, B), B, 0, stream>>>(s, N_NODES * N_IN);
    k_fill1  <<<blocks(N_NODES, B), B, 0, stream>>>(deg, cnt, N_NODES);

    // weight conversion f32 -> f16
    k_cvt_f32_f16<<<blocks(N_H1 * N_IN, B), B, 0, stream>>>(Wc, Wch, N_H1 * N_IN);
    k_cvt_f32_f16<<<blocks(N_H2 * N_H1, B), B, 0, stream>>>(W1, W1h, N_H2 * N_H1);
    k_cvt_f32_f16<<<blocks(N_H3 * N_H2, B), B, 0, stream>>>(W2, W2h, N_H3 * N_H2);

    // aggregation
    k_degcnt  <<<blocks(N_EDGES, B), B, 0, stream>>>(row, col, deg, cnt);
    k_msg     <<<blocks((long long)N_EDGES * 32, B), B, 0, stream>>>(x, row, col, deg, s);
    k_finalize<<<blocks((long long)N_NODES * (N_IN / 4), B), B, 0, stream>>>(s, x, deg, cnt, aggh);

    // MLP via WMMA (N_NODES = 6250 * 16 exactly, no edge masks)
    const int mTiles = N_NODES / 16;       // 6250
    k_wmma_gemm<N_IN, N_H1, true ><<<mTiles, (N_H1 / 64) * 32, 0, stream>>>(aggh, Wch, bc, h1);
    k_wmma_gemm<N_H1, N_H2, true ><<<mTiles, (N_H2 / 64) * 32, 0, stream>>>(h1,   W1h, b1, h2);
    k_wmma_gemm<N_H2, N_H3, true ><<<mTiles, (N_H3 / 64) * 32, 0, stream>>>(h2,   W2h, b2, h3);

    // final projection to 1 output
    k_final<<<blocks(N_NODES, B), B, 0, stream>>>(h3, W3, b3, out);
}
